// CrossAttention_21174188769831
// MI455X (gfx1250) — compile-verified, matrix operand order checked by reading, not run on hardware
//
#include <hip/hip_runtime.h>
#include <hip/hip_bf16.h>

typedef __attribute__((ext_vector_type(16))) __bf16 v16bf;
typedef __attribute__((ext_vector_type(8)))  __bf16 v8bf;
typedef __attribute__((ext_vector_type(8)))  float  v8f;
typedef __attribute__((ext_vector_type(4)))  float  v4f;
typedef unsigned int u32x4 __attribute__((ext_vector_type(4)));
typedef int          i32x4 __attribute__((ext_vector_type(4)));
typedef int          i32x8 __attribute__((ext_vector_type(8)));

#define C_DIM 768           // K and N of every GEMM
#define BM 128
#define BN 128
#define BK 32
#define NSTEPS (C_DIM / BK) // 24
#define LDS_STRIDE 40       // padded row stride (elements) -> 80B rows, conflict-free b128 reads

// ---------------------------------------------------------------------------
// f32 -> bf16 conversion (weights prep; tiny)
// ---------------------------------------------------------------------------
__global__ void cvt_f32_bf16_kernel(const float* __restrict__ src,
                                    __bf16* __restrict__ dst, int n) {
    int i = blockIdx.x * blockDim.x + threadIdx.x;
    int stride = gridDim.x * blockDim.x;
    for (; i < n; i += stride) dst[i] = (__bf16)src[i];
}

static __device__ inline v16bf mk16(v8bf lo, v8bf hi) {
    v16bf r;
#pragma unroll
    for (int e = 0; e < 8; ++e) { r[e] = lo[e]; r[8 + e] = hi[e]; }
    return r;
}

// ---------------------------------------------------------------------------
// TDM: DMA a 128x32 bf16 tile (row stride 768 elems) from global into LDS,
// with LDS padding: after every 16 DWORDs (one 64B row) insert 4 DWORDs (16B)
// -> padded LDS row stride of 40 elements, matching the compute layout.
// Descriptor per CDNA5 ISA ch.8 (D# groups 0/1; groups 2/3 zero for 2D).
// ---------------------------------------------------------------------------
static __device__ inline void tdm_load_tile_bf16(const __bf16* gsrc, unsigned lds_off) {
    unsigned long long ga = (unsigned long long)(size_t)gsrc;
    u32x4 g0;
    g0[0] = 1u;                                                // count=1, gather off
    g0[1] = lds_off;                                           // lds_addr (bytes)
    g0[2] = (unsigned)(ga & 0xffffffffu);                      // global_addr[31:0]
    g0[3] = (unsigned)((ga >> 32) & 0x01ffffffu) | (2u << 30); // ga[56:32] | type=2
    i32x8 g1;
    g1[0] = (int)((1u << 16)      // data_size = 2 bytes
                | (1u << 20)      // pad_enable
                | (3u << 22)      // pad_interval: 16 DWORDs
                | (3u << 25));    // pad_amount: 4 DWORDs
    g1[1] = (int)((unsigned)C_DIM << 16);  // tensor_dim0[15:0] in [31:16]
    g1[2] = (int)((unsigned)C_DIM << 16);  // dim0 hi=0 | tensor_dim1[15:0] in [31:16]
    g1[3] = (int)((unsigned)BK << 16);     // dim1 hi=0 | tile_dim0=32 in [31:16]
    g1[4] = BM;                            // tile_dim1=128, tile_dim2=0
    g1[5] = C_DIM;                         // tensor_dim0_stride low32 (elements)
    g1[6] = 0;
    g1[7] = 0;
    i32x4 z4 = (i32x4)0;
#if __clang_major__ >= 23
    i32x8 z8 = (i32x8)0;
    __builtin_amdgcn_tensor_load_to_lds(g0, g1, z4, z4, z8, 0);
#else
    __builtin_amdgcn_tensor_load_to_lds(g0, g1, z4, z4, 0);
#endif
}

// ---------------------------------------------------------------------------
// D[M x 768] = A[M x 768] * W^T (+ bias), bf16 WMMA, f32 accumulate.
// A_IS_F32: A is f32 (converted to bf16 on the VALU while staging to LDS);
//           else A is bf16 and its tiles are DMA'd by the TDM.
// B (weight) tiles are always DMA'd by the TDM. Ping-pong LDS buffers:
// one barrier per K-step; stage(s+1) overlaps WMMA(s).
// Block = 256 threads = 8 waves; tile 128x128; each wave 64x32 (4x2 wmma).
// ---------------------------------------------------------------------------
template <bool A_IS_F32, bool OUT_F32>
__global__ __launch_bounds__(256)
void gemm_wmma_kernel(const void* __restrict__ Av,
                      const __bf16* __restrict__ W,
                      void* __restrict__ Dv,
                      const float* __restrict__ bias) {
    __shared__ __bf16 As[2][BM * LDS_STRIDE];
    __shared__ __bf16 Bs[2][BN * LDS_STRIDE];

    const int tid   = threadIdx.x;
    const int lane  = tid & 31;
    const int wave  = tid >> 5;
    const int waveM = wave & 1;    // 2 waves along M: 64 rows each
    const int waveN = wave >> 1;   // 4 waves along N: 32 cols each

    const int tileM = blockIdx.x * BM;
    const int tileN = blockIdx.y * BN;

    // staging (VALU path): each of 256 threads copies one 16-element half-row
    const int cpRow = tid >> 1;
    const int cpCol = (tid & 1) * 16;

    // fragment addressing (ISA 16-bit A layout): lanes 0-15 -> K[0..7],K[16..23]
    const int lhalf = lane >> 4;
    const int lrow  = lane & 15;
    const int kfrag = lhalf * 8;

    // stage A tile via VALU (f32 -> bf16) into buffer `buf` for K-offset `kk`
    auto stageA_f32 = [&](int buf, int kk) {
        const float* A   = (const float*)Av;
        const float* src = A + (size_t)(tileM + cpRow) * C_DIM + kk + cpCol;
        if (kk + BK < C_DIM) __builtin_prefetch(src + BK, 0, 1);
        v4f f0 = *(const v4f*)(src);
        v4f f1 = *(const v4f*)(src + 4);
        v4f f2 = *(const v4f*)(src + 8);
        v4f f3 = *(const v4f*)(src + 12);
        v8bf h0, h1;
#pragma unroll
        for (int e = 0; e < 4; ++e) {
            h0[e]     = (__bf16)f0[e];
            h0[4 + e] = (__bf16)f1[e];
            h1[e]     = (__bf16)f2[e];
            h1[4 + e] = (__bf16)f3[e];
        }
        __bf16* d = &As[buf][cpRow * LDS_STRIDE + cpCol];
        *(v8bf*)(d)     = h0;
        *(v8bf*)(d + 8) = h1;
    };

    // issue TDM DMA(s) for buffer `buf`, K-offset `kk` (one wave only)
    auto stageDMA = [&](int buf, int kk) {
        if (wave == 0) {
            tdm_load_tile_bf16(W + (size_t)tileN * C_DIM + kk,
                               (unsigned)(size_t)(void*)&Bs[buf][0]);
            if (!A_IS_F32)
                tdm_load_tile_bf16((const __bf16*)Av + (size_t)tileM * C_DIM + kk,
                                   (unsigned)(size_t)(void*)&As[buf][0]);
        }
    };

    v8f acc[4][2];
#pragma unroll
    for (int i = 0; i < 4; ++i)
#pragma unroll
        for (int j = 0; j < 2; ++j) acc[i][j] = (v8f)0.0f;

    // ---- prologue: fill buffer 0 ----
    if (A_IS_F32) stageA_f32(0, 0);
    stageDMA(0, 0);
    __builtin_amdgcn_s_wait_tensorcnt(0);
    __syncthreads();

    for (int s = 0; s < NSTEPS; ++s) {
        const int cur = s & 1;
        const int nxt = cur ^ 1;

        // stage next tile while computing current
        if (s + 1 < NSTEPS) {
            if (A_IS_F32) stageA_f32(nxt, (s + 1) * BK);
            stageDMA(nxt, (s + 1) * BK);
        }

        v16bf aF[4], bF[2];
#pragma unroll
        for (int i = 0; i < 4; ++i) {
            const __bf16* p = &As[cur][(waveM * 64 + i * 16 + lrow) * LDS_STRIDE + kfrag];
            aF[i] = mk16(*(const v8bf*)p, *(const v8bf*)(p + 16));
        }
#pragma unroll
        for (int j = 0; j < 2; ++j) {
            const __bf16* p = &Bs[cur][(waveN * 32 + j * 16 + lrow) * LDS_STRIDE + kfrag];
            bF[j] = mk16(*(const v8bf*)p, *(const v8bf*)(p + 16));
        }

#pragma unroll
        for (int i = 0; i < 4; ++i)
#pragma unroll
            for (int j = 0; j < 2; ++j)
                acc[i][j] = __builtin_amdgcn_wmma_f32_16x16x32_bf16(
                    false, aF[i], false, bF[j], (short)0, acc[i][j], false, false);

        // next-tile DMA done (no-op for non-issuing waves), then release buffers
        __builtin_amdgcn_s_wait_tensorcnt(0);
        __syncthreads();
    }

    // ---- epilogue: C/D layout: VGPR r -> M = r + (laneHi ? 8 : 0), N = lane&15
    const int nCol    = lane & 15;
    const int rowHalf = lhalf * 8;
#pragma unroll
    for (int i = 0; i < 4; ++i) {
#pragma unroll
        for (int j = 0; j < 2; ++j) {
            const int col   = tileN + waveN * 32 + j * 16 + nCol;
            const int rbase = tileM + waveM * 64 + i * 16 + rowHalf;
#pragma unroll
            for (int r = 0; r < 8; ++r) {
                float v = acc[i][j][r];
                if (OUT_F32) {
                    float* D = (float*)Dv;
                    D[(size_t)(rbase + r) * C_DIM + col] = v + bias[col];
                } else {
                    __bf16* D = (__bf16*)Dv;
                    D[(size_t)(rbase + r) * C_DIM + col] = (__bf16)v;
                }
            }
        }
    }
}

// ---------------------------------------------------------------------------
// Per-token 2-way attention combine. One wave32 per (token, head).
// 96 = 32 lanes * 3 elements. Two shuffle-reduced dots, softmax over 2.
// ---------------------------------------------------------------------------
__global__ __launch_bounds__(256)
void attn_combine_kernel(const __bf16* __restrict__ Q,
                         const __bf16* __restrict__ Kp,
                         const __bf16* __restrict__ Vp,
                         __bf16* __restrict__ X) {
    const int wave  = blockIdx.x * (blockDim.x >> 5) + (threadIdx.x >> 5);
    const int lane  = threadIdx.x & 31;
    const int head  = wave & 7;                 // NH = 8
    const int token = wave >> 3;                // 0 .. 32767

    const size_t qoff = (size_t)token * C_DIM + head * 96;
    const size_t off0 = ((size_t)token * 2 + 0) * C_DIM + head * 96;
    const size_t off1 = ((size_t)token * 2 + 1) * C_DIM + head * 96;
    const int d0 = lane * 3;

    float q[3], k0[3], k1[3];
#pragma unroll
    for (int e = 0; e < 3; ++e) {
        q[e]  = (float)Q[qoff + d0 + e];
        k0[e] = (float)Kp[off0 + d0 + e];
        k1[e] = (float)Kp[off1 + d0 + e];
    }
    float l0 = q[0] * k0[0] + q[1] * k0[1] + q[2] * k0[2];
    float l1 = q[0] * k1[0] + q[1] * k1[1] + q[2] * k1[2];
#pragma unroll
    for (int off = 16; off; off >>= 1) {
        l0 += __shfl_xor(l0, off, 32);
        l1 += __shfl_xor(l1, off, 32);
    }
    const float scale = 0.1020620726159658f;    // 96^-0.5
    l0 *= scale; l1 *= scale;
    const float m  = fmaxf(l0, l1);
    const float e0 = __expf(l0 - m);
    const float e1 = __expf(l1 - m);
    const float inv = 1.0f / (e0 + e1);
    const float a0 = e0 * inv, a1 = e1 * inv;

#pragma unroll
    for (int e = 0; e < 3; ++e) {
        float v0 = (float)Vp[off0 + d0 + e];
        float v1 = (float)Vp[off1 + d0 + e];
        X[qoff + d0 + e] = (__bf16)(a0 * v0 + a1 * v1);
    }
}

// ---------------------------------------------------------------------------
// Launcher
// ---------------------------------------------------------------------------
extern "C" void kernel_launch(void* const* d_in, const int* in_sizes, int n_in,
                              void* d_out, int out_size, void* d_ws, size_t ws_size,
                              hipStream_t stream) {
    const float* query = (const float*)d_in[0];   // [B,N,C]      = [32768, 768]
    const float* key   = (const float*)d_in[1];   // [B,N,2,C]    = [65536, 768]
    const float* value = (const float*)d_in[2];   // [B,N,2,C]
    const float* Wq    = (const float*)d_in[3];   // [768,768]
    const float* Wk    = (const float*)d_in[4];   // [768,768]  (applied to key AND value)
    const float* Wp    = (const float*)d_in[5];   // [768,768]
    const float* bp    = (const float*)d_in[6];   // [768]
    float* out = (float*)d_out;

    const int Mq  = in_sizes[0] / C_DIM;          // 32768 tokens
    const int Mkv = in_sizes[1] / C_DIM;          // 65536 projected rows

    // workspace carve-up (all bf16): 3 weights + Q + K + V + X  (~306 MB)
    char* ws = (char*)d_ws;
    const size_t WSZ = (size_t)C_DIM * C_DIM * sizeof(__bf16);
    __bf16* Wq_bf = (__bf16*)ws;  ws += WSZ;
    __bf16* Wk_bf = (__bf16*)ws;  ws += WSZ;
    __bf16* Wp_bf = (__bf16*)ws;  ws += WSZ;
    __bf16* Qbf   = (__bf16*)ws;  ws += (size_t)Mq  * C_DIM * sizeof(__bf16);
    __bf16* Kbf   = (__bf16*)ws;  ws += (size_t)Mkv * C_DIM * sizeof(__bf16);
    __bf16* Vbf   = (__bf16*)ws;  ws += (size_t)Mkv * C_DIM * sizeof(__bf16);
    __bf16* Xbf   = (__bf16*)ws;  ws += (size_t)Mq  * C_DIM * sizeof(__bf16);

    // 1) weights -> bf16 (L2-resident afterwards)
    const int wn = C_DIM * C_DIM;
    cvt_f32_bf16_kernel<<<(wn + 255) / 256, 256, 0, stream>>>(Wq, Wq_bf, wn);
    cvt_f32_bf16_kernel<<<(wn + 255) / 256, 256, 0, stream>>>(Wk, Wk_bf, wn);
    cvt_f32_bf16_kernel<<<(wn + 255) / 256, 256, 0, stream>>>(Wp, Wp_bf, wn);

    // 2) projections (f32 activations converted on the fly, bf16 out; TDM weights)
    dim3 blk(256);
    dim3 gq(Mq / BM, C_DIM / BN);
    dim3 gkv(Mkv / BM, C_DIM / BN);
    gemm_wmma_kernel<true, false><<<gq,  blk, 0, stream>>>(query, Wq_bf, Qbf, nullptr);
    gemm_wmma_kernel<true, false><<<gkv, blk, 0, stream>>>(key,   Wk_bf, Kbf, nullptr);
    gemm_wmma_kernel<true, false><<<gkv, blk, 0, stream>>>(value, Wk_bf, Vbf, nullptr);

    // 3) per-token softmax over 2 views + combine -> X (bf16)
    const int totalWaves = Mq * 8;                 // tokens * heads
    attn_combine_kernel<<<totalWaves / 8, 256, 0, stream>>>(Qbf, Kbf, Vbf, Xbf);

    // 4) output projection: bf16 A via TDM, f32 + bias out
    gemm_wmma_kernel<false, true><<<gq, blk, 0, stream>>>(Xbf, Wp_bf, out, bp);
}
